// PeepholeLSTMCell_14920716386505
// MI455X (gfx1250) — matrix operand order repaired and checked by reading, not compiled, not run on hardware
//
#include <hip/hip_runtime.h>
#include <hip/hip_bf16.h>

// ---------------------------------------------------------------------------
// Peephole LSTM cell, MI455X (gfx1250, wave32, WMMA).
//   B=8192, D=U=1024.  z = [x|h] @ [W;Uw]  (K=2048, N=4096), fused epilogue.
//   GEMM in bf16 via v_wmma_f32_16x16x32_bf16; z never hits memory.
//   Wave tile: 64(M) x 32(U) x 4 gates -> 32 wmma / K-step, 24 b128 loads.
// ---------------------------------------------------------------------------

#define LSTM_B 8192
#define LSTM_U 1024
#define LSTM_K2 2048   // D + U
#define LSTM_N4 4096   // 4 * U

typedef __bf16 bf16_t;
typedef __attribute__((ext_vector_type(8)))  __bf16 v8bf;
typedef __attribute__((ext_vector_type(16))) __bf16 v16bf;
typedef __attribute__((ext_vector_type(8)))  float  v8f;
typedef __attribute__((ext_vector_type(8)))  unsigned short us8;

__device__ __forceinline__ unsigned short f32_to_bf16_rne(float f) {
    unsigned int u = __builtin_bit_cast(unsigned int, f);
    unsigned int r = u + 0x7FFFu + ((u >> 16) & 1u);
    return (unsigned short)(r >> 16);
}

__device__ __forceinline__ v16bf cat8(v8bf lo, v8bf hi) {
    return __builtin_shufflevector(lo, hi, 0, 1, 2, 3, 4, 5, 6, 7,
                                           8, 9, 10, 11, 12, 13, 14, 15);
}

__device__ __forceinline__ float sigmoidf_fast(float x) {
    return 1.0f / (1.0f + __expf(-x));
}

// ---------------------------------------------------------------------------
// Pack [x | h] (f32) -> xh bf16 [B][K2], 8 elements per thread (b128 I/O).
// ---------------------------------------------------------------------------
__global__ void lstm_pack_xh(const float* __restrict__ x,
                             const float* __restrict__ h,
                             unsigned short* __restrict__ xh) {
    size_t t   = (size_t)blockIdx.x * blockDim.x + threadIdx.x;
    size_t idx = t * 8;                       // flat index into [B][K2]
    int row = (int)(idx >> 11);               // / 2048
    int k   = (int)(idx & 2047);
    const float* src = (k < LSTM_U) ? (x + (size_t)row * LSTM_U + k)
                                    : (h + (size_t)row * LSTM_U + (k - LSTM_U));
    const float4* s4 = (const float4*)src;
    float4 a = s4[0];
    float4 b = s4[1];
    us8 o;
    o[0] = f32_to_bf16_rne(a.x); o[1] = f32_to_bf16_rne(a.y);
    o[2] = f32_to_bf16_rne(a.z); o[3] = f32_to_bf16_rne(a.w);
    o[4] = f32_to_bf16_rne(b.x); o[5] = f32_to_bf16_rne(b.y);
    o[6] = f32_to_bf16_rne(b.z); o[7] = f32_to_bf16_rne(b.w);
    *(us8*)(xh + idx) = o;
}

// ---------------------------------------------------------------------------
// Build wt bf16 [N4][K2]: wt[n][k] = Wcat[k][n], n = gate*1024 + u,
// Wcat = [[Wi Wf Wc Wo],[Ui Uf Uc Uo]].  LDS-tiled 32x32 transpose.
// ---------------------------------------------------------------------------
__global__ void lstm_pack_wt(const float* __restrict__ Wi, const float* __restrict__ Wf,
                             const float* __restrict__ Wc, const float* __restrict__ Wo,
                             const float* __restrict__ Ui, const float* __restrict__ Uf,
                             const float* __restrict__ Uc, const float* __restrict__ Uo,
                             unsigned short* __restrict__ wt) {
    __shared__ float tile[32][33];
    int tx = threadIdx.x, ty = threadIdx.y;
    int k0 = blockIdx.x * 32;                 // 0..2047
    int n0 = blockIdx.y * 32;                 // 0..4095
    int gate = n0 >> 10;
    int u0   = n0 & 1023;
    bool isU = (k0 >= LSTM_U);
    int kk0  = isU ? (k0 - LSTM_U) : k0;
    const float* src =
        isU ? (gate == 0 ? Ui : gate == 1 ? Uf : gate == 2 ? Uc : Uo)
            : (gate == 0 ? Wi : gate == 1 ? Wf : gate == 2 ? Wc : Wo);
    tile[ty][tx] = src[(size_t)(kk0 + ty) * LSTM_U + u0 + tx];
    __syncthreads();
    wt[(size_t)(n0 + ty) * LSTM_K2 + k0 + tx] = f32_to_bf16_rne(tile[tx][ty]);
}

// ---------------------------------------------------------------------------
// Fused GEMM + peephole epilogue.
// 128 threads = 4 waves.  All 4 waves of a block share the same 32 u-columns
// (identical B fragments -> WGP$ hits) and take 4 consecutive 64-row M tiles.
// Per wave: acc[4 gates][4 mi][2 ni] v8f = 256 VGPRs of accumulators.
// Per K-step (32): 32 v_wmma_f32_16x16x32_bf16, 8 A-loads + 16 B-loads (b128).
// ---------------------------------------------------------------------------
__global__ __launch_bounds__(128, 1)
void lstm_gemm_cell(const unsigned short* __restrict__ xh_u16,
                    const unsigned short* __restrict__ wt_u16,
                    const float* __restrict__ cin,
                    const float* __restrict__ Vi, const float* __restrict__ Vf,
                    const float* __restrict__ Vo,
                    const float* __restrict__ bi, const float* __restrict__ bff,
                    const float* __restrict__ bc, const float* __restrict__ bo,
                    float* __restrict__ out) {
    const bf16_t* xh = (const bf16_t*)xh_u16;
    const bf16_t* wt = (const bf16_t*)wt_u16;

    int wib   = (int)threadIdx.x >> 5;        // wave-in-block 0..3
    int lane  = (int)threadIdx.x & 31;
    int l15   = lane & 15;
    int lhi   = lane >> 4;                    // 0 or 1
    int utile = (int)blockIdx.x & 31;         // shared by all 4 waves
    int mtile = ((int)blockIdx.x >> 5) * 4 + wib;   // 0..127
    int M0 = mtile * 64;
    int U0 = utile * 32;

    v8f acc[4][4][2];                         // [gate][mi][ni]
    const v8f vzero = {0.f, 0.f, 0.f, 0.f, 0.f, 0.f, 0.f, 0.f};
#pragma unroll
    for (int g = 0; g < 4; ++g)
#pragma unroll
        for (int mi = 0; mi < 4; ++mi)
#pragma unroll
            for (int ni = 0; ni < 2; ++ni) acc[g][mi][ni] = vzero;

    // K loop: 2048 in steps of 32, 32 WMMAs per step.
    for (int kk = 0; kk < LSTM_K2; kk += 32) {
        v16bf afrag[4];
#pragma unroll
        for (int mi = 0; mi < 4; ++mi) {
            const bf16_t* p =
                xh + (size_t)(M0 + mi * 16 + l15) * LSTM_K2 + kk + lhi * 8;
            v8bf lo = *(const v8bf*)p;          // K: kblk + [0..7]
            v8bf hi = *(const v8bf*)(p + 16);   // K: kblk + 16 + [0..7]
            afrag[mi] = cat8(lo, hi);
        }
#pragma unroll
        for (int g = 0; g < 4; ++g) {
#pragma unroll
            for (int ni = 0; ni < 2; ++ni) {
                const bf16_t* q =
                    wt + (size_t)(g * LSTM_U + U0 + ni * 16 + l15) * LSTM_K2 +
                    kk + lhi * 8;
                v8bf lo = *(const v8bf*)q;       // K: kblk + [0..7]
                v8bf hi = *(const v8bf*)(q + 16);
                v16bf bfrag = cat8(lo, hi);
#pragma unroll
                for (int mi = 0; mi < 4; ++mi) {
                    acc[g][mi][ni] = __builtin_amdgcn_wmma_f32_16x16x32_bf16(
                        false, afrag[mi], false, bfrag, (short)0,
                        acc[g][mi][ni], false, false);
                }
            }
        }
    }

    // Epilogue: C/D layout -> lane l15 = N, component j = M row (+8 if lhi).
    const size_t cplane = (size_t)LSTM_B * LSTM_U;
#pragma unroll
    for (int ni = 0; ni < 2; ++ni) {
        int u = U0 + ni * 16 + l15;
        float vi = Vi[u], vf = Vf[u], vo = Vo[u];
        float bbi = bi[u], bbf = bff[u], bbc = bc[u], bbo = bo[u];
#pragma unroll
        for (int mi = 0; mi < 4; ++mi) {
#pragma unroll
            for (int j = 0; j < 8; ++j) {
                int r = M0 + mi * 16 + lhi * 8 + j;
                float c  = cin[(size_t)r * LSTM_U + u];
                float zi = acc[0][mi][ni][j] + bbi;
                float zf = acc[1][mi][ni][j] + bbf;
                float zc = acc[2][mi][ni][j] + bbc;
                float zo = acc[3][mi][ni][j] + bbo;
                float it = sigmoidf_fast(zi + vi * c);
                float ft = sigmoidf_fast(zf + vf * c);
                float ct = ft * c + it * tanhf(zc);
                float ot = sigmoidf_fast(zo + vo * ct);
                float ht = ot * tanhf(ct);
                out[(size_t)r * LSTM_U + u]          = ht;
                out[cplane + (size_t)r * LSTM_U + u] = ct;
            }
        }
    }
}

// ---------------------------------------------------------------------------
// Launch
// ---------------------------------------------------------------------------
extern "C" void kernel_launch(void* const* d_in, const int* in_sizes, int n_in,
                              void* d_out, int out_size, void* d_ws, size_t ws_size,
                              hipStream_t stream) {
    const float* x  = (const float*)d_in[0];
    const float* h  = (const float*)d_in[1];
    const float* c  = (const float*)d_in[2];
    const float* Wi = (const float*)d_in[3];
    const float* Wf = (const float*)d_in[4];
    const float* Wc = (const float*)d_in[5];
    const float* Wo = (const float*)d_in[6];
    const float* Ui = (const float*)d_in[7];
    const float* Uf = (const float*)d_in[8];
    const float* Uc = (const float*)d_in[9];
    const float* Uo = (const float*)d_in[10];
    const float* Vi = (const float*)d_in[11];
    const float* Vf = (const float*)d_in[12];
    const float* Vo = (const float*)d_in[13];
    const float* bi = (const float*)d_in[14];
    const float* bf = (const float*)d_in[15];
    const float* bc = (const float*)d_in[16];
    const float* bo = (const float*)d_in[17];
    float* out = (float*)d_out;

    unsigned short* xh = (unsigned short*)d_ws;                       // 32 MB
    unsigned short* wt = xh + (size_t)LSTM_B * LSTM_K2;               // 16 MB

    // 1) pack [x|h] -> bf16
    {
        int total = LSTM_B * LSTM_K2 / 8;     // threads
        lstm_pack_xh<<<total / 256, 256, 0, stream>>>(x, h, xh);
    }
    // 2) pack + transpose weights -> bf16 [N4][K2]
    {
        dim3 grid(LSTM_K2 / 32, LSTM_N4 / 32);
        dim3 blk(32, 32);
        lstm_pack_wt<<<grid, blk, 0, stream>>>(Wi, Wf, Wc, Wo, Ui, Uf, Uc, Uo, wt);
    }
    // 3) fused GEMM + LSTM cell
    {
        // (B/64) * (U/32) = 128 * 32 = 4096 waves -> 1024 blocks of 4 waves
        int blocks = (LSTM_B / 64) * (LSTM_U / 32) / 4;
        lstm_gemm_cell<<<blocks, 128, 0, stream>>>(xh, wt, c, Vi, Vf, Vo,
                                                   bi, bf, bc, bo, out);
    }
}
